// ContextExchangeBlock_42786464203259
// MI455X (gfx1250) — matrix operand order repaired and checked by reading, not compiled, not run on hardware
//
#include <hip/hip_runtime.h>
#include <hip/hip_bf16.h>
#include <cstdint>

// ---------------------------------------------------------------------------
// Sizes (match reference)
//   D=1024, H=16, HD=64, S=4, B=8, L=2048, HID=4096
//   NCHUNK=8 chunks of 256 rows per (s,b)  -> 256 attention workgroups
// ---------------------------------------------------------------------------

typedef __attribute__((ext_vector_type(16))) __bf16 v16bf;
typedef __attribute__((ext_vector_type(8)))  float  v8f;

#define DEV __device__ __forceinline__

DEV float wred(float v) {
  v += __shfl_xor(v, 16);
  v += __shfl_xor(v, 8);
  v += __shfl_xor(v, 4);
  v += __shfl_xor(v, 2);
  v += __shfl_xor(v, 1);
  return v;
}

// 16-bit A-matrix (16xK=32) half-index -> K mapping (ISA 7.12.2):
// lanes 0-15: v0..v3 K=0..7, v4..v7 K=16..23 ; lanes 16-31: +8
DEV int kmapA(int j, int hi) { return ((j & 8) << 1) + hi * 8 + (j & 7); }

DEV unsigned short bf16bits(float f) {
  return __builtin_bit_cast(unsigned short, (__bf16)f);
}
DEV __bf16 us2bf(unsigned short u) { return __builtin_bit_cast(__bf16, u); }
DEV float sigmoidf_(float x) { return 1.0f / (1.0f + __expf(-x)); }

// Async-copy one 64KB tile (16 rows x 1024 f32) global -> LDS.
// 256 threads x 16B x 16 iterations; per wave: 16 async b128 ops (ASYNCcnt+=16).
DEV void async_copy_64k(uint32_t lds_off, const float* gsrc, int tid) {
  uint32_t dst = lds_off + (uint32_t)tid * 16u;
  uint64_t src = (uint64_t)(uintptr_t)((const char*)gsrc + tid * 16);
#pragma unroll
  for (int i = 0; i < 16; ++i) {
    asm volatile("global_load_async_to_lds_b128 %0, %1, off"
                 :: "v"(dst), "v"(src) : "memory");
    dst += 4096u;
    src += 4096u;
  }
}

// ---------------------------------------------------------------------------
// q_in = rmsnorm(context) * q_norm_w      (1 block, wave b per row)
// ---------------------------------------------------------------------------
__global__ __launch_bounds__(256) void ceb_qin(
    const float* __restrict__ context, const float* __restrict__ q_norm_w,
    float* __restrict__ q_in) {
  int tid = threadIdx.x, wv = tid >> 5, lane = tid & 31;
  const float* cp = context + wv * 1024;
  float xv[32]; float ss = 0.f;
#pragma unroll
  for (int i = 0; i < 32; ++i) { xv[i] = cp[lane + (i << 5)]; ss += xv[i] * xv[i]; }
  ss = wred(ss);
  float rms = rsqrtf(ss * (1.0f / 1024.0f) + 1e-6f);
#pragma unroll
  for (int i = 0; i < 32; ++i) {
    int d = lane + (i << 5);
    q_in[wv * 1024 + d] = xv[i] * rms * q_norm_w[d];
  }
}

// ---------------------------------------------------------------------------
// q[b,i] = q_in[b]·Wq[i] + bq[i]   (wave per output, 1024 blocks x 8 waves)
// ---------------------------------------------------------------------------
__global__ __launch_bounds__(256) void ceb_qproj(
    const float* __restrict__ q_in, const float* __restrict__ in_proj_w,
    const float* __restrict__ in_proj_b, float* __restrict__ qbuf) {
  int tid = threadIdx.x, wv = tid >> 5, lane = tid & 31;
  int o = blockIdx.x * 8 + wv;      // 0..8191
  int b = o >> 10, i = o & 1023;
  const float* wr = in_proj_w + (size_t)i * 1024;
  const float* qr = q_in + b * 1024;
  float acc = 0.f;
#pragma unroll
  for (int k = 0; k < 32; ++k) { int d = lane + (k << 5); acc += qr[d] * wr[d]; }
  acc = wred(acc);
  if (lane == 0) qbuf[o] = acc + in_proj_b[i];
}

// ---------------------------------------------------------------------------
// wq_eff[b,h,d] = (1/8) * sum_j q[b,h*64+j] * Wk[h*64+j, d]  -> bf16
// thread per output, 512 blocks x 256
// ---------------------------------------------------------------------------
__global__ __launch_bounds__(256) void ceb_wqeff(
    const float* __restrict__ qbuf, const float* __restrict__ in_proj_w,
    unsigned short* __restrict__ wq_us) {
  int o = blockIdx.x * 256 + threadIdx.x;   // 0..131071
  int b = o >> 14, rem = o & 16383;
  int h = rem >> 10, d = rem & 1023;
  const float* qr = qbuf + (b << 10) + (h << 6);
  const float* wk = in_proj_w + ((size_t)(1024 + (h << 6)) << 10) + d;  // row-major [3D,D]
  float acc = 0.f;
#pragma unroll 8
  for (int j = 0; j < 64; ++j) acc += qr[j] * wk[(size_t)j << 10];
  wq_us[o] = bf16bits(acc * 0.125f);        // fold 1/sqrt(64)
}

// ---------------------------------------------------------------------------
// Attention chunk kernel: flash-decode over a 256-row chunk of one (s,b).
//   Async double-buffered 64KB tile stream (global -> LDS, ASYNCcnt pipelined).
//   per 16-row tile: rmsnorm -> bf16 x_hat in LDS
//     scores[16l x 16h] = x_hat @ wq_eff^T   (WMMA bf16, K split across waves)
//     online softmax (per head) -> p (bf16), alpha
//     z[16h x 1024d] = alpha*z + p^T @ x_hat (WMMA bf16, 64 N-tiles / 8 waves)
//   partial (z, m, lsum) -> workspace
// ---------------------------------------------------------------------------
__global__ __launch_bounds__(256) void ceb_attn_chunk(
    const float* __restrict__ seq, const unsigned char* __restrict__ masks,
    const float* __restrict__ s_norm_w, const unsigned short* __restrict__ wq_us,
    float* __restrict__ part) {
  extern __shared__ unsigned char smraw[];
  float* raw    = (float*)smraw;            // 2 x 16384 f : raw tile ring
  float* z      = raw + 32768;              // 16384 f : z[h][d]
  float* snw    = z + 16384;                // 1024 f
  float* spart  = snw + 1024;               // 2048 f : [wave][l][h]
  float* sc     = spart + 2048;             // 256 f  : [l][h]
  float* m_s    = sc + 256;                 // 16
  float* lsum_s = m_s + 16;                 // 16
  float* alphab = lsum_s + 16;              // 16
  __bf16* xh    = (__bf16*)(alphab + 16);   // 16384 h : x_hat[l][d]
  __bf16* wqs   = xh + 16384;               // 16384 h : wq_eff[h][d]
  __bf16* pbuf  = wqs + 16384;              // 512 h   : p^T [h][K=32]
  unsigned char* maskrow = (unsigned char*)(pbuf + 512);  // 16 B
  // LDS byte offset of `raw` (dynamic LDS starts after static allocations)
  const uint32_t ldsRawBase = __builtin_amdgcn_groupstaticsize();

  const int NCHUNK = 8;
  int blk = blockIdx.x;
  int sb = blk / NCHUNK, ck = blk % NCHUNK;
  int s = sb >> 3, b = sb & 7;
  int tid = threadIdx.x, wv = tid >> 5, lane = tid & 31;
  int row16 = lane & 15, hi = lane >> 4;

  // init
  for (int i = tid; i < 16384; i += 256) z[i] = 0.f;
  for (int i = tid; i < 16384; i += 256) wqs[i] = us2bf(wq_us[((size_t)b << 14) + i]);
  for (int i = tid; i < 1024; i += 256) snw[i] = s_norm_w[i];
  for (int i = tid; i < 512; i += 256) pbuf[i] = (__bf16)0.f;
  if (tid < 16) { m_s[tid] = -3.0e38f; lsum_s[tid] = 0.f; }

  const size_t seqbase = (((size_t)s * 8 + b) * 2048) * 1024;
  const unsigned char* maskp = masks + ((size_t)s * 8 + b) * 2048;
  const float* tbase = seq + seqbase + (size_t)ck * 256 * 1024;

  // prologue: tile 0 -> buffer 0
  async_copy_64k(ldsRawBase, tbase, tid);
  __syncthreads();   // init visible to all

  for (int t = 0; t < 16; ++t) {
    int cur = t & 1;
    // issue next tile into the other buffer, then wait for current tile
    if (t + 1 < 16) {
      async_copy_64k(ldsRawBase + (uint32_t)(cur ^ 1) * 65536u,
                     tbase + (size_t)(t + 1) * 16384, tid);
      asm volatile("s_wait_asynccnt 0x10" ::: "memory");
    } else {
      asm volatile("s_wait_asynccnt 0x0" ::: "memory");
    }
    __syncthreads();   // all waves' copy slices for tile t landed

    int gl0 = ck * 256 + t * 16;
    const float* rb = raw + cur * 16384;
    // --- rmsnorm 16 rows from LDS raw tile, write bf16 x_hat (2 rows/wave) ---
    for (int rr = 0; rr < 2; ++rr) {
      int r = (wv << 1) + rr;
      float xv[32]; float ss = 0.f;
#pragma unroll
      for (int i = 0; i < 32; ++i) {
        xv[i] = rb[(r << 10) + lane + (i << 5)];
        ss += xv[i] * xv[i];
      }
      ss = wred(ss);
      float rms = rsqrtf(ss * (1.0f / 1024.0f) + 1e-6f);
#pragma unroll
      for (int i = 0; i < 32; ++i) {
        int d = lane + (i << 5);
        xh[(r << 10) + d] = (__bf16)(xv[i] * rms * snw[d]);
      }
    }
    if (tid < 16) maskrow[tid] = maskp[gl0 + tid];
    __syncthreads();

    // --- scores: K split across 8 waves (128 each = 4 WMMA) ---
    {
      v8f c;
#pragma unroll
      for (int g = 0; g < 8; ++g) c[g] = 0.f;
      int k0 = wv << 7;
#pragma unroll
      for (int ts = 0; ts < 4; ++ts) {
        int kb = k0 + (ts << 5);
        v16bf A, Bv;
#pragma unroll
        for (int j = 0; j < 16; ++j) A[j] = xh[(row16 << 10) + kb + kmapA(j, hi)];
#pragma unroll
        for (int j = 0; j < 16; ++j) Bv[j] = wqs[(row16 << 10) + kb + kmapA(j, hi)];
        c = __builtin_amdgcn_wmma_f32_16x16x32_bf16(false, A, false, Bv,
                                                    (short)0, c, false, false);
      }
#pragma unroll
      for (int g = 0; g < 8; ++g)
        spart[(wv << 8) + ((g + hi * 8) << 4) + row16] = c[g];
    }
    __syncthreads();

    // --- reduce partials + mask (thread = (l,h)) ---
    {
      float ssum = 0.f;
#pragma unroll
      for (int w = 0; w < 8; ++w) ssum += spart[(w << 8) + tid];
      sc[tid] = maskrow[tid >> 4] ? -3.0e38f : ssum;
    }
    __syncthreads();

    // --- online softmax per head ---
    if (tid < 16) {
      float mm = m_s[tid], tmax = mm;
      for (int l = 0; l < 16; ++l) tmax = fmaxf(tmax, sc[(l << 4) + tid]);
      float alpha = __expf(mm - tmax);
      float psum = 0.f;
      for (int l = 0; l < 16; ++l) {
        float p = (tmax <= -1.0e37f) ? 0.f : __expf(sc[(l << 4) + tid] - tmax);
        pbuf[(tid << 5) + l] = (__bf16)p;
        psum += p;
      }
      lsum_s[tid] = lsum_s[tid] * alpha + psum;
      m_s[tid] = tmax;
      alphab[tid] = alpha;
    }
    __syncthreads();

    // --- z = alpha*z + p^T @ x_hat : 8 N-tiles of 16 per wave ---
    for (int i = 0; i < 8; ++i) {
      int n0 = ((wv << 3) + i) << 4;
      v16bf A, Bv; v8f c;
#pragma unroll
      for (int j = 0; j < 16; ++j) A[j] = pbuf[(row16 << 5) + kmapA(j, hi)];
#pragma unroll
      for (int j = 0; j < 16; ++j) {
        int K = kmapA(j, hi);                   // K>=16 pairs with A==0
        Bv[j] = xh[((K & 15) << 10) + n0 + row16];
      }
#pragma unroll
      for (int g = 0; g < 8; ++g) {
        int m = g + hi * 8;
        c[g] = z[(m << 10) + n0 + row16] * alphab[m];
      }
      c = __builtin_amdgcn_wmma_f32_16x16x32_bf16(false, A, false, Bv,
                                                  (short)0, c, false, false);
#pragma unroll
      for (int g = 0; g < 8; ++g) z[((g + hi * 8) << 10) + n0 + row16] = c[g];
    }
    __syncthreads();
  }

  // --- write partial (z[16][1024], m[16], lsum[16]) ---
  size_t base = (size_t)(sb * NCHUNK + ck) * 16416;
  for (int i = tid; i < 16384; i += 256) part[base + i] = z[i];
  if (tid < 16) {
    part[base + 16384 + tid] = m_s[tid];
    part[base + 16400 + tid] = lsum_s[tid];
  }
}

// ---------------------------------------------------------------------------
// Combine chunks + V/out projection start:
//   attended[s,b,o] = znorm[h]·Wv[h*64+j] + bv      (32 blocks)
// ---------------------------------------------------------------------------
__global__ __launch_bounds__(256) void ceb_combine(
    const float* __restrict__ part, const float* __restrict__ in_proj_w,
    const float* __restrict__ in_proj_b, float* __restrict__ attended) {
  __shared__ float Mh[16], Wc[128], Ls[16];
  extern __shared__ float zn[];             // 16*1024
  int sb = blockIdx.x;
  int tid = threadIdx.x, wv = tid >> 5, lane = tid & 31;

  if (tid < 16) {
    float M = -3.0e38f;
    for (int c = 0; c < 8; ++c)
      M = fmaxf(M, part[(size_t)(sb * 8 + c) * 16416 + 16384 + tid]);
    float L = 0.f;
    for (int c = 0; c < 8; ++c) {
      size_t pb = (size_t)(sb * 8 + c) * 16416;
      float w = __expf(part[pb + 16384 + tid] - M);
      Wc[c * 16 + tid] = w;
      L += w * part[pb + 16400 + tid];
    }
    Mh[tid] = M; Ls[tid] = L;
  }
  __syncthreads();
  for (int idx = tid; idx < 16384; idx += 256) {
    int h = idx >> 10;
    float acc = 0.f;
    for (int c = 0; c < 8; ++c)
      acc += Wc[c * 16 + h] * part[(size_t)(sb * 8 + c) * 16416 + idx];
    zn[idx] = acc / Ls[h];
  }
  __syncthreads();
  for (int o = wv; o < 1024; o += 8) {
    int h = o >> 6;
    const float* wr = in_proj_w + ((size_t)(2048 + o) << 10);
    float acc = 0.f;
#pragma unroll
    for (int k = 0; k < 32; ++k) { int d = lane + (k << 5); acc += wr[d] * zn[(h << 10) + d]; }
    acc = wred(acc);
    if (lane == 0) attended[(sb << 10) + o] = acc + in_proj_b[2048 + o];
  }
}

// ---------------------------------------------------------------------------
// seq_ctx[b,i] = out_b[i] + (1/4) sum_s attended[s,b]·out_w[i]
// ---------------------------------------------------------------------------
__global__ __launch_bounds__(256) void ceb_seqctx(
    const float* __restrict__ attended, const float* __restrict__ out_w,
    const float* __restrict__ out_b, float* __restrict__ seq_ctx) {
  int tid = threadIdx.x, wv = tid >> 5, lane = tid & 31;
  int o = blockIdx.x * 8 + wv;
  int b = o >> 10, i = o & 1023;
  const float* wr = out_w + (size_t)i * 1024;
  float acc = 0.f;
#pragma unroll
  for (int k = 0; k < 32; ++k) {
    int d = lane + (k << 5);
    float a = attended[(b << 10) + d] + attended[((8 + b) << 10) + d] +
              attended[((16 + b) << 10) + d] + attended[((24 + b) << 10) + d];
    acc += wr[d] * a;
  }
  acc = wred(acc);
  if (lane == 0) seq_ctx[o] = 0.25f * acc + out_b[i];
}

// ---------------------------------------------------------------------------
// gate + residual: ctx = context + sigmoid([context,seq_ctx]·gw[i]+gb)*seq_ctx
// ---------------------------------------------------------------------------
__global__ __launch_bounds__(256) void ceb_gate(
    const float* __restrict__ context, const float* __restrict__ seq_ctx,
    const float* __restrict__ gate_w, const float* __restrict__ gate_b,
    float* __restrict__ ctxb) {
  int tid = threadIdx.x, wv = tid >> 5, lane = tid & 31;
  int o = blockIdx.x * 8 + wv;
  int b = o >> 10, i = o & 1023;
  const float* wr = gate_w + (size_t)i * 2048;
  float acc = 0.f;
#pragma unroll
  for (int k = 0; k < 64; ++k) {
    int d = lane + (k << 5);
    float src = (d < 1024) ? context[(b << 10) + d] : seq_ctx[(b << 10) + d - 1024];
    acc += wr[d] * src;
  }
  acc = wred(acc);
  if (lane == 0) {
    float g = sigmoidf_(acc + gate_b[i]);
    ctxb[o] = context[o] + g * seq_ctx[o];
  }
}

// ---------------------------------------------------------------------------
// h = rmsnorm(ctx) * ffn_norm_w     (1 block, wave per row)
// ---------------------------------------------------------------------------
__global__ __launch_bounds__(256) void ceb_hnorm(
    const float* __restrict__ ctxb, const float* __restrict__ ffn_norm_w,
    float* __restrict__ hbuf) {
  int tid = threadIdx.x, wv = tid >> 5, lane = tid & 31;
  const float* cp = ctxb + wv * 1024;
  float xv[32]; float ss = 0.f;
#pragma unroll
  for (int i = 0; i < 32; ++i) { xv[i] = cp[lane + (i << 5)]; ss += xv[i] * xv[i]; }
  ss = wred(ss);
  float rms = rsqrtf(ss * (1.0f / 1024.0f) + 1e-6f);
#pragma unroll
  for (int i = 0; i < 32; ++i) {
    int d = lane + (i << 5);
    hbuf[wv * 1024 + d] = xv[i] * rms * ffn_norm_w[d];
  }
}

// ---------------------------------------------------------------------------
// act[b,j] = silu(h·W_g[j]+b_g) * (h·W_v[j]+b_v)    (4096 blocks)
// ---------------------------------------------------------------------------
__global__ __launch_bounds__(256) void ceb_gateup(
    const float* __restrict__ hbuf, const float* __restrict__ gate_up_w,
    const float* __restrict__ gate_up_b, float* __restrict__ act) {
  int tid = threadIdx.x, wv = tid >> 5, lane = tid & 31;
  int o = blockIdx.x * 8 + wv;
  int b = o >> 12, j = o & 4095;
  const float* wg = gate_up_w + (size_t)j * 1024;
  const float* wvp = gate_up_w + (size_t)(4096 + j) * 1024;
  const float* hr = hbuf + (b << 10);
  float ag = 0.f, av = 0.f;
#pragma unroll
  for (int k = 0; k < 32; ++k) {
    int d = lane + (k << 5);
    float h = hr[d];
    ag += h * wg[d];
    av += h * wvp[d];
  }
  ag = wred(ag); av = wred(av);
  if (lane == 0) {
    float g = ag + gate_up_b[j];
    float v = av + gate_up_b[4096 + j];
    act[o] = (g * sigmoidf_(g)) * v;
  }
}

// ---------------------------------------------------------------------------
// out[b,i] = ctx[b,i] + act[b]·down_w[i] + down_b[i]   (1024 blocks)
// ---------------------------------------------------------------------------
__global__ __launch_bounds__(256) void ceb_down(
    const float* __restrict__ ctxb, const float* __restrict__ act,
    const float* __restrict__ down_w, const float* __restrict__ down_b,
    float* __restrict__ out) {
  int tid = threadIdx.x, wv = tid >> 5, lane = tid & 31;
  int o = blockIdx.x * 8 + wv;
  int b = o >> 10, i = o & 1023;
  const float* wr = down_w + (size_t)i * 4096;
  const float* ar = act + (b << 12);
  float acc = 0.f;
#pragma unroll 16
  for (int k = 0; k < 128; ++k) { int d = lane + (k << 5); acc += ar[d] * wr[d]; }
  acc = wred(acc);
  if (lane == 0) out[o] = ctxb[o] + acc + down_b[i];
}

// ---------------------------------------------------------------------------
extern "C" void kernel_launch(void* const* d_in, const int* in_sizes, int n_in,
                              void* d_out, int out_size, void* d_ws, size_t ws_size,
                              hipStream_t stream) {
  (void)in_sizes; (void)n_in; (void)out_size; (void)ws_size;
  const float* context   = (const float*)d_in[0];
  const float* sequences = (const float*)d_in[1];
  const float* q_norm_w  = (const float*)d_in[2];
  const float* s_norm_w  = (const float*)d_in[3];
  const float* in_proj_w = (const float*)d_in[4];
  const float* in_proj_b = (const float*)d_in[5];
  const float* out_w     = (const float*)d_in[6];
  const float* out_b     = (const float*)d_in[7];
  const float* gate_w    = (const float*)d_in[8];
  const float* gate_b    = (const float*)d_in[9];
  const float* ffn_norm_w= (const float*)d_in[10];
  const float* gate_up_w = (const float*)d_in[11];
  const float* gate_up_b = (const float*)d_in[12];
  const float* down_w    = (const float*)d_in[13];
  const float* down_b    = (const float*)d_in[14];
  const unsigned char* masks = (const unsigned char*)d_in[15];
  float* out = (float*)d_out;

  // workspace carve (floats)
  float* f_ws     = (float*)d_ws;
  float* q_in     = f_ws;                     // 8192
  float* qbuf     = q_in + 8192;              // 8192
  float* part     = qbuf + 8192;              // 32*8*16416 = 4202496
  float* attended = part + 4202496;           // 32768
  float* seq_ctx  = attended + 32768;         // 8192
  float* ctxb     = seq_ctx + 8192;           // 8192
  float* hbuf     = ctxb + 8192;              // 8192
  float* act      = hbuf + 8192;              // 32768
  unsigned short* wq_us = (unsigned short*)(act + 32768);  // 131072 bf16

  // attn LDS: raw ring 131072B + f32 arrays + bf16 arrays + mask
  const size_t ATTN_SMEM =
      (32768 + 16384 + 1024 + 2048 + 256 + 16 + 16 + 16) * sizeof(float) +
      (16384 + 16384 + 512) * 2 + 32;
  const size_t COMB_SMEM = 16384 * sizeof(float);

  ceb_qin<<<1, 256, 0, stream>>>(context, q_norm_w, q_in);
  ceb_qproj<<<1024, 256, 0, stream>>>(q_in, in_proj_w, in_proj_b, qbuf);
  ceb_wqeff<<<512, 256, 0, stream>>>(qbuf, in_proj_w, wq_us);
  ceb_attn_chunk<<<256, 256, ATTN_SMEM, stream>>>(sequences, masks, s_norm_w,
                                                  wq_us, part);
  ceb_combine<<<32, 256, COMB_SMEM, stream>>>(part, in_proj_w, in_proj_b,
                                              attended);
  ceb_seqctx<<<1024, 256, 0, stream>>>(attended, out_w, out_b, seq_ctx);
  ceb_gate<<<1024, 256, 0, stream>>>(context, seq_ctx, gate_w, gate_b, ctxb);
  ceb_hnorm<<<1, 256, 0, stream>>>(ctxb, ffn_norm_w, hbuf);
  ceb_gateup<<<4096, 256, 0, stream>>>(hbuf, gate_up_w, gate_up_b, act);
  ceb_down<<<1024, 256, 0, stream>>>(ctxb, act, down_w, down_b, out);
}
